// HungarianMatcher_3908420239659
// MI455X (gfx1250) — matrix operand order repaired and checked by reading, not compiled, not run on hardware
//
#include <hip/hip_runtime.h>
#include <math.h>

typedef __attribute__((ext_vector_type(16))) _Float16 v16h;
typedef __attribute__((ext_vector_type(8)))  _Float16 v8h;
typedef __attribute__((ext_vector_type(8)))  float    v8f;

#define C_CLS 92
#define C_PAD 96
#define T_TOT 1600
#define NTJ   (T_TOT / 16)   // 100 t-tiles

// ---------------------------------------------------------------------------
// Kernel 1: row softmax over [BQ, 92] logits -> f16 probs padded to [BQ, 96].
// One wave (32 lanes) per row; 8 rows per 256-thread block.
// ---------------------------------------------------------------------------
__global__ __launch_bounds__(256) void softmax_f16_kernel(
    const float* __restrict__ logits, _Float16* __restrict__ prob, int rows) {
  int lane = threadIdx.x & 31;
  int row  = blockIdx.x * 8 + (threadIdx.x >> 5);
  if (row >= rows) return;
  const float* src = logits + (size_t)row * C_CLS;

  float x0 = src[lane];                                         // cols 0..31
  float x1 = src[lane + 32];                                    // cols 32..63
  float x2 = (lane + 64 < C_CLS) ? src[lane + 64] : -INFINITY;  // cols 64..91

  float m = fmaxf(fmaxf(x0, x1), x2);
  #pragma unroll
  for (int off = 16; off > 0; off >>= 1) m = fmaxf(m, __shfl_xor(m, off, 32));

  float e0 = __expf(x0 - m);
  float e1 = __expf(x1 - m);
  float e2 = __expf(x2 - m);          // 0 for padded lanes (x2 = -inf)
  float s  = e0 + e1 + e2;
  #pragma unroll
  for (int off = 16; off > 0; off >>= 1) s += __shfl_xor(s, off, 32);
  float inv = 1.0f / s;

  _Float16* dst = prob + (size_t)row * C_PAD;
  dst[lane]      = (_Float16)(e0 * inv);
  dst[lane + 32] = (_Float16)(e1 * inv);
  dst[lane + 64] = (_Float16)(e2 * inv);  // cols 92..95 naturally 0
}

// ---------------------------------------------------------------------------
// Kernel 2: materialize one-hot WMMA B fragments for every t-tile.
// bfrag[(tj*3 + c)*32 + lane] in contiguous k order (same as the A loads):
//   lane (n = lane&15, half = lane>>4) holds B[K = c*32 + 16*half + j, n]
// 100 tiles x 3 chunks x 32 lanes x 32 B = 300 KB (L2-resident).
// ---------------------------------------------------------------------------
__global__ __launch_bounds__(256) void build_bfrag_kernel(
    const int* __restrict__ labels, v16h* __restrict__ bfrag) {
  int g = blockIdx.x * 256 + threadIdx.x;
  if (g >= NTJ * 32) return;
  int tj = g >> 5, lane = g & 31;
  int nl = lane & 15, half = lane >> 4;
  int lab = labels[tj * 16 + nl];
  #pragma unroll
  for (int c = 0; c < 3; ++c) {
    int rel = lab - (c * 32 + 16 * half);
    v16h Bf;
    #pragma unroll
    for (int j = 0; j < 16; ++j)
      Bf[j] = (rel == j) ? (_Float16)1.0f : (_Float16)0.0f;
    bfrag[(tj * 3 + c) * 32 + lane] = Bf;
  }
}

// ---------------------------------------------------------------------------
// Kernel 3: fused cost matrix.
//   cost_class via WMMA f32<=f16 : prob[16x96] @ onehot(labels)[96x16]
//   cost_bbox / cost_giou via f32 VALU epilogue (v_rcp_f32 fast divides).
// One block of 8 waves per 16-query tile; each wave sweeps t-tiles with
// software-pipelined (double-buffered) B-fragment loads so the L2 latency
// hides under the previous tile's epilogue.
// ---------------------------------------------------------------------------
__global__ __launch_bounds__(256) void cost_kernel(
    const float*    __restrict__ pred_boxes,  // [BQ,4] cxcywh
    const float*    __restrict__ tgt_boxes,   // [T,4] cxcywh
    const _Float16* __restrict__ prob,        // [BQ,96] f16
    const v16h*     __restrict__ bfrag,       // [NTJ][3][32] one-hot B frags
    float*          __restrict__ out)         // [BQ,T]
{
  // cxcywh(4) + xyxy(4) + area(1) = 9 floats per box
  __shared__ float s_tgt[T_TOT][9];   // 57600 B
  __shared__ float s_pred[16][9];     //   576 B   (total < 64 KB)

  const int tid = threadIdx.x;
  const int i0  = blockIdx.x * 16;

  // ---- stage all targets ----
  for (int t = tid; t < T_TOT; t += 256) {
    float cx = tgt_boxes[t * 4 + 0], cy = tgt_boxes[t * 4 + 1];
    float w  = tgt_boxes[t * 4 + 2], h  = tgt_boxes[t * 4 + 3];
    s_tgt[t][0] = cx; s_tgt[t][1] = cy; s_tgt[t][2] = w; s_tgt[t][3] = h;
    s_tgt[t][4] = cx - 0.5f * w; s_tgt[t][5] = cy - 0.5f * h;
    s_tgt[t][6] = cx + 0.5f * w; s_tgt[t][7] = cy + 0.5f * h;
    s_tgt[t][8] = w * h;
  }
  // ---- stage this block's 16 pred boxes ----
  if (tid < 16) {
    int i = i0 + tid;
    float cx = pred_boxes[i * 4 + 0], cy = pred_boxes[i * 4 + 1];
    float w  = pred_boxes[i * 4 + 2], h  = pred_boxes[i * 4 + 3];
    s_pred[tid][0] = cx; s_pred[tid][1] = cy; s_pred[tid][2] = w; s_pred[tid][3] = h;
    s_pred[tid][4] = cx - 0.5f * w; s_pred[tid][5] = cy - 0.5f * h;
    s_pred[tid][6] = cx + 0.5f * w; s_pred[tid][7] = cy + 0.5f * h;
    s_pred[tid][8] = w * h;
  }
  __syncthreads();

  const int lane = tid & 31;
  const int wid  = tid >> 5;
  const int nl   = lane & 15;   // A: row m index | B/C: column n index
  const int half = lane >> 4;

  // ---- load A fragments once (row m = lane&15, contiguous k = c*32 + 16*half + j, same order as B) ----
  const _Float16* arow = prob + (size_t)(i0 + nl) * C_PAD;
  v16h A[3];
  #pragma unroll
  for (int c = 0; c < 3; ++c) {
    A[c] = *(const v16h*)(arow + c * 32 + 16 * half);
  }

  // ---- software-pipelined sweep over t-tiles (wave-uniform control flow,
  //      EXEC stays all-ones for WMMA) ----
  const v16h* bfL = bfrag + lane;
  v16h b0 = bfL[(wid * 3 + 0) * 32];
  v16h b1 = bfL[(wid * 3 + 1) * 32];
  v16h b2 = bfL[(wid * 3 + 2) * 32];

  for (int tj = wid; tj < NTJ; tj += 8) {
    // prefetch next tile's B fragments (clamped, wave-uniform)
    const int tjn = (tj + 8 < NTJ) ? tj + 8 : tj;
    const v16h n0 = bfL[(tjn * 3 + 0) * 32];
    const v16h n1 = bfL[(tjn * 3 + 1) * 32];
    const v16h n2 = bfL[(tjn * 3 + 2) * 32];

    v8f acc = {};
    acc = __builtin_amdgcn_wmma_f32_16x16x32_f16(false, A[0], false, b0,
                                                 (short)0, acc, false, false);
    acc = __builtin_amdgcn_wmma_f32_16x16x32_f16(false, A[1], false, b1,
                                                 (short)0, acc, false, false);
    acc = __builtin_amdgcn_wmma_f32_16x16x32_f16(false, A[2], false, b2,
                                                 (short)0, acc, false, false);

    const int t = tj * 16 + nl;       // this lane's target column

    // target operands (broadcast LDS reads across the two 16-lane halves)
    const float tc0 = s_tgt[t][0], tc1 = s_tgt[t][1];
    const float tc2 = s_tgt[t][2], tc3 = s_tgt[t][3];
    const float tx0 = s_tgt[t][4], ty0 = s_tgt[t][5];
    const float tx1 = s_tgt[t][6], ty1 = s_tgt[t][7];
    const float ta  = s_tgt[t][8];

    #pragma unroll
    for (int r = 0; r < 8; ++r) {
      const int il = r + 8 * half;    // WMMA C layout: M = r + 8*(lane/16)
      const float pc0 = s_pred[il][0], pc1 = s_pred[il][1];
      const float pc2 = s_pred[il][2], pc3 = s_pred[il][3];
      const float px0 = s_pred[il][4], py0 = s_pred[il][5];
      const float px1 = s_pred[il][6], py1 = s_pred[il][7];
      const float pa  = s_pred[il][8];

      // L1 on cxcywh
      const float l1 = fabsf(pc0 - tc0) + fabsf(pc1 - tc1) +
                       fabsf(pc2 - tc2) + fabsf(pc3 - tc3);

      // GIoU on xyxy (fast reciprocals: v_rcp_f32, ~1 ulp)
      const float iw    = fmaxf(fminf(px1, tx1) - fmaxf(px0, tx0), 0.0f);
      const float ih    = fmaxf(fminf(py1, ty1) - fmaxf(py0, ty0), 0.0f);
      const float inter = iw * ih;
      const float uni   = pa + ta - inter;
      const float ew    = fmaxf(fmaxf(px1, tx1) - fminf(px0, tx0), 0.0f);
      const float eh    = fmaxf(fmaxf(py1, ty1) - fminf(py0, ty0), 0.0f);
      const float ae    = ew * eh;
      const float giou  = inter * __builtin_amdgcn_rcpf(uni) -
                          (ae - uni) * __builtin_amdgcn_rcpf(ae);

      // cost = 5*L1 + 1*(-prob[i,label]) + 2*(-giou); streaming NT store
      __builtin_nontemporal_store(5.0f * l1 - acc[r] - 2.0f * giou,
                                  &out[(size_t)(i0 + il) * T_TOT + t]);
    }

    // rotate double buffer (compiler places the s_wait_loadcnt here,
    // after the epilogue, instead of right before the WMMAs)
    b0 = n0; b1 = n1; b2 = n2;
  }
}

// ---------------------------------------------------------------------------
extern "C" void kernel_launch(void* const* d_in, const int* in_sizes, int n_in,
                              void* d_out, int out_size, void* d_ws, size_t ws_size,
                              hipStream_t stream) {
  const float* pred_logits = (const float*)d_in[0];  // [B,Q,92]
  const float* pred_boxes  = (const float*)d_in[1];  // [B,Q,4]
  const int*   tgt_labels  = (const int*)d_in[2];    // [T]
  const float* tgt_boxes   = (const float*)d_in[3];  // [T,4]
  float*       out         = (float*)d_out;          // [B,Q,T]

  const int BQ = in_sizes[1] / 4;                    // 14400

  _Float16* prob = (_Float16*)d_ws;                  // [BQ,96] f16 = 2.76 MB
  size_t prob_bytes = ((size_t)BQ * C_PAD * sizeof(_Float16) + 255) & ~(size_t)255;
  v16h* bfrag = (v16h*)((char*)d_ws + prob_bytes);   // 300 KB

  softmax_f16_kernel<<<(BQ + 7) / 8, 256, 0, stream>>>(pred_logits, prob, BQ);
  build_bfrag_kernel<<<(NTJ * 32 + 255) / 256, 256, 0, stream>>>(tgt_labels, bfrag);
  cost_kernel<<<BQ / 16, 256, 0, stream>>>(pred_boxes, tgt_boxes, prob, bfrag, out);
}
